// GTS_37306085933703
// MI455X (gfx1250) — matrix operand order repaired and checked by reading, not compiled, not run on hardware
//
#include <hip/hip_runtime.h>
#include <hip/hip_bf16.h>

// ---------------------------------------------------------------------------
// DCRNN encoder/decoder for MI455X (gfx1250).
// Heavy math runs through v_wmma_f32_16x16x32_bf16 (bf16 in, f32 accumulate).
// Chebyshev GEMM: 2x2 register tiling per wave + B-panel staged in LDS
// (shared by the 4 waves of the workgroup);
// B fragments come from conflict-free ds_load_b128.
// ---------------------------------------------------------------------------

typedef __attribute__((ext_vector_type(16))) __bf16 v16bf;
typedef __attribute__((ext_vector_type(8)))  __bf16 bf16x8;
typedef __attribute__((ext_vector_type(8)))  float  v8f;
typedef __attribute__((ext_vector_type(4)))  float  f32x4;

#define B_     32
#define T_     12
#define N_     512
#define H_     64
#define D_     65          // D_IN + H (== D_OUT + H)
#define M_     4           // k_hop + 1
#define C_     (D_ * B_)   // 2080 columns of the diffusion GEMM
#define R_     (B_ * N_)   // 16384 rows of the linear GEMM
#define DM_    260         // D_ * M_
#define DMP_   288         // DM_ padded to multiple of 32 (K of linear GEMM)
#define NPRED_ 12

// LDS column stride for the staged B panel: 520 halves = 1040 B.
// 1040 mod 256 == 16, so the 16 lanes of a half-wave hit 16 distinct
// 4-bank groups -> conflict-free ds_load_b128.
#define BSH_STRIDE 520

__device__ __forceinline__ float sigmoidf_(float x) {
    return 1.0f / (1.0f + __expf(-x));
}

// ---------------------------------------------------------------------------
// Chebyshev diffusion GEMM:  Y = alpha * (A @ X) + beta * Xsub
//   A   : bf16 [512][512] row-major
//   X   : bf16 transposed [C_][512]   (Xt[c][k] = X[k][c])
//   Y   : fp32 + bf16, transposed [C_][512]
// Block = 4 waves; block output = 128 rows x 32 cols.
// Each wave: 2 M-tiles x 2 N-tiles (4 accumulators, 4 WMMAs per k-step).
// B panel (32 X-columns x full K=512) staged in LDS once, shared by waves.
// grid (4, 65), block (32, 4). 16 k-steps of K=32.
// ---------------------------------------------------------------------------
__global__ __launch_bounds__(128) void cheb_gemm(
    const __bf16* __restrict__ Abf,
    const __bf16* __restrict__ Xbf,
    const float*  __restrict__ Xsub,
    float alpha, float beta,
    float*  __restrict__ Yt,
    __bf16* __restrict__ Ybf)
{
    __shared__ __bf16 Bsh[32 * BSH_STRIDE];

    const int lane = threadIdx.x;          // 0..31
    const int wid  = threadIdx.y;          // 0..3
    const int tid  = wid * 32 + lane;      // 0..127
    const int tn0  = blockIdx.y * 2;       // block's N-tile pair (0..129)

    // ---- cooperative stage of B panel: 32 columns x 512 K halves ----
    {
        const int cl = tid & 31;           // column within panel
        const int k0 = (tid >> 5) * 128;   // K quarter
        const __bf16* src = Xbf + (size_t)(tn0 * 16 + cl) * 512 + k0;
        __bf16* dst = &Bsh[cl * BSH_STRIDE + k0];
        #pragma unroll
        for (int i = 0; i < 16; ++i)
            *(bf16x8*)(dst + i * 8) = *(const bf16x8*)(src + i * 8);
    }
    __syncthreads();

    const int m   = lane & 15;
    const int hi  = lane >> 4;                     // 0 or 1
    const int tm0 = blockIdx.x * 8 + wid * 2;      // wave's first M tile (0..30)

    v8f acc00 = {}, acc01 = {}, acc10 = {}, acc11 = {};

    // A fragment: lane = row, per lane K = {koff..koff+7} and {koff+16..koff+23}
    const __bf16* paB0 = Abf + (size_t)(tm0 * 16 + m) * 512 + hi * 8;
    const __bf16* paB1 = paB0 + (size_t)16 * 512;
    // B fragment (LDS): lane = column, per lane K = {koff..koff+7} and {koff+16..koff+23}
    const __bf16* pbB0 = &Bsh[(0 * 16 + m) * BSH_STRIDE + hi * 8];
    const __bf16* pbB1 = &Bsh[(1 * 16 + m) * BSH_STRIDE + hi * 8];

    for (int kk = 0; kk < 16; ++kk) {
        union { bf16x8 h[2]; v16bf v; } a0, a1, b0, b1;
        const __bf16* pa0 = paB0 + kk * 32;
        const __bf16* pa1 = paB1 + kk * 32;
        a0.h[0] = *(const bf16x8*)pa0;  a0.h[1] = *(const bf16x8*)(pa0 + 16);
        a1.h[0] = *(const bf16x8*)pa1;  a1.h[1] = *(const bf16x8*)(pa1 + 16);
        const __bf16* pb0 = pbB0 + kk * 32;
        const __bf16* pb1 = pbB1 + kk * 32;
        b0.h[0] = *(const bf16x8*)pb0;  b0.h[1] = *(const bf16x8*)(pb0 + 16);
        b1.h[0] = *(const bf16x8*)pb1;  b1.h[1] = *(const bf16x8*)(pb1 + 16);
        acc00 = __builtin_amdgcn_wmma_f32_16x16x32_bf16(
                    false, a0.v, false, b0.v, (short)0, acc00, false, false);
        acc01 = __builtin_amdgcn_wmma_f32_16x16x32_bf16(
                    false, a0.v, false, b1.v, (short)0, acc01, false, false);
        acc10 = __builtin_amdgcn_wmma_f32_16x16x32_bf16(
                    false, a1.v, false, b0.v, (short)0, acc10, false, false);
        acc11 = __builtin_amdgcn_wmma_f32_16x16x32_bf16(
                    false, a1.v, false, b1.v, (short)0, acc11, false, false);
    }

    // ---- fused combine + dual-precision store of the 4 output tiles ----
    #pragma unroll
    for (int i = 0; i < 2; ++i) {
        const int rowbase = (tm0 + i) * 16 + hi * 8;   // 8 consecutive rows
        #pragma unroll
        for (int j = 0; j < 2; ++j) {
            v8f acc = (i == 0) ? (j == 0 ? acc00 : acc01)
                               : (j == 0 ? acc10 : acc11);
            const int col = (tn0 + j) * 16 + m;
            float*  py  = Yt  + (size_t)col * 512 + rowbase;
            __bf16* pyb = Ybf + (size_t)col * 512 + rowbase;
            float sub[8];
            if (beta != 0.0f) {
                const float* ps = Xsub + (size_t)col * 512 + rowbase;
                f32x4 s0 = *(const f32x4*)ps;
                f32x4 s1 = *(const f32x4*)(ps + 4);
                #pragma unroll
                for (int v = 0; v < 4; ++v) { sub[v] = s0[v]; sub[4 + v] = s1[v]; }
            }
            f32x4 o0, o1; bf16x8 ob;
            #pragma unroll
            for (int v = 0; v < 8; ++v) {
                float val = alpha * acc[v];
                if (beta != 0.0f) val += beta * sub[v];
                if (v < 4) o0[v] = val; else o1[v - 4] = val;
                ob[v] = (__bf16)val;
            }
            *(f32x4*)py       = o0;
            *(f32x4*)(py + 4) = o1;
            *(bf16x8*)pyb     = ob;
        }
    }
}

// ---------------------------------------------------------------------------
// Linear projection GEMM:  Outt = (XK @ W)^T
//   XK  : bf16 [R_][288] row-major
//   Wt  : bf16 [nout][288]  (W transposed, K zero-padded)
//   Outt: fp32 [nout][R_]
// grid (256, nout/16), block (32,4). 9 k-steps of K=32.
// ---------------------------------------------------------------------------
__global__ __launch_bounds__(128) void lin_gemm(
    const __bf16* __restrict__ XK,
    const __bf16* __restrict__ Wt,
    float* __restrict__ Outt)
{
    const int lane = threadIdx.x;
    const int m    = lane & 15;
    const int hi   = lane >> 4;
    const int tm   = blockIdx.x * 4 + threadIdx.y;   // 0..1023
    const int tn   = blockIdx.y;

    v8f acc = {};
    const __bf16* paB = XK + (size_t)(tm * 16 + m) * DMP_ + hi * 8;
    const __bf16* pbB = Wt + (size_t)(tn * 16 + m) * DMP_ + hi * 8;

    for (int kk = 0; kk < DMP_ / 32; ++kk) {
        union { bf16x8 h[2]; v16bf v; } a, b;
        const __bf16* pa = paB + kk * 32;
        const __bf16* pb = pbB + kk * 32;
        a.h[0] = *(const bf16x8*)pa;
        a.h[1] = *(const bf16x8*)(pa + 16);
        b.h[0] = *(const bf16x8*)pb;
        b.h[1] = *(const bf16x8*)(pb + 16);
        acc = __builtin_amdgcn_wmma_f32_16x16x32_bf16(
                  false, a.v, false, b.v, (short)0, acc, false, false);
    }

    const int col     = tn * 16 + m;
    const int rowbase = tm * 16 + hi * 8;
    float* py = Outt + (size_t)col * R_ + rowbase;
    f32x4 o0, o1;
    #pragma unroll
    for (int v = 0; v < 4; ++v) { o0[v] = acc[v]; o1[v] = acc[4 + v]; }
    *(f32x4*)py       = o0;
    *(f32x4*)(py + 4) = o1;
}

// ---------------------------------------------------------------------------
// Pack X0 = concat(x, (gate?)sigmoid(ru_r+b)*h : h) into transposed layout.
//   X0t[c][n], c = d*B_ + b ; d==0 -> x input, d>0 -> hidden feature d-1.
// ---------------------------------------------------------------------------
__global__ void pack_x(const float* __restrict__ xsrc, int xstrideB,
                       const float* __restrict__ Ht,
                       const float* __restrict__ gate,     // RUt rows 0..63 or null
                       const float* __restrict__ bru,
                       float* __restrict__ X0t, __bf16* __restrict__ X0bf)
{
    int idx = blockIdx.x * blockDim.x + threadIdx.x;
    if (idx >= C_ * N_) return;
    const int n = idx & (N_ - 1);
    const int c = idx >> 9;
    const int b = c & (B_ - 1);
    const int d = c >> 5;
    float val;
    if (d == 0) {
        val = xsrc[b * xstrideB + n];
    } else {
        const int r = b * N_ + n;
        float hv = Ht[(size_t)(d - 1) * R_ + r];
        if (gate) {
            float g = sigmoidf_(gate[(size_t)(d - 1) * R_ + r] + bru[d - 1]);
            hv *= g;
        }
        val = hv;
    }
    X0t[idx]  = val;
    X0bf[idx] = (__bf16)val;
}

// Gather the 4 diffusion panels into xk rows: XK[r][d*4+m] (K padded to 288).
__global__ void gather_xk(const float* __restrict__ XPAN,   // [4][C_][512] fp32
                          __bf16* __restrict__ XK)           // [R_][288]
{
    int idx = blockIdx.x * blockDim.x + threadIdx.x;
    if (idx >= R_ * DMP_) return;
    const int dm = idx % DMP_;
    const int r  = idx / DMP_;
    __bf16 v = (__bf16)0.0f;
    if (dm < DM_) {
        const int d  = dm >> 2;
        const int mm = dm & 3;
        const int b  = r >> 9;
        const int n  = r & (N_ - 1);
        v = (__bf16)XPAN[((size_t)mm * C_ + d * B_ + b) * N_ + n];
    }
    XK[idx] = v;
}

// GRU state update: h = u*h + (1-u)*tanh(c), u = sigmoid(ru_u + b).
__global__ void update_h(float* __restrict__ Ht,
                         const float* __restrict__ RUt, const float* __restrict__ bru,
                         const float* __restrict__ Cct, const float* __restrict__ bc)
{
    int idx = blockIdx.x * blockDim.x + threadIdx.x;
    if (idx >= H_ * R_) return;
    const int f = idx >> 14;          // feature
    const int r = idx & (R_ - 1);
    float u = sigmoidf_(RUt[(size_t)(H_ + f) * R_ + r] + bru[H_ + f]);
    float c = tanhf(Cct[(size_t)f * R_ + r] + bc[f]);
    Ht[idx] = u * Ht[idx] + (1.0f - u) * c;
}

// Decoder output projection: out = h @ Wo + bo; feeds next step + d_out[:,t].
__global__ void project_out(const float* __restrict__ Ht,
                            const float* __restrict__ Wo, const float* __restrict__ bo,
                            float* __restrict__ dec_in, float* __restrict__ dout, int t)
{
    int r = blockIdx.x * blockDim.x + threadIdx.x;
    if (r >= R_) return;
    float s = bo[0];
    #pragma unroll
    for (int f = 0; f < H_; ++f) s += Ht[(size_t)f * R_ + r] * Wo[f];
    dec_in[r] = s;
    const int b = r >> 9;
    const int n = r & (N_ - 1);
    dout[((size_t)b * NPRED_ + t) * N_ + n] = s;
}

__global__ void convert_graph(const float* __restrict__ A, __bf16* __restrict__ Abf, int n)
{
    int i = blockIdx.x * blockDim.x + threadIdx.x;
    if (i < n) Abf[i] = (__bf16)A[i];
}

// W fp32 [260][nout] -> Wt bf16 [nout][288] zero-padded.
__global__ void convert_w(const float* __restrict__ W, __bf16* __restrict__ Wt, int nout)
{
    int idx = blockIdx.x * blockDim.x + threadIdx.x;
    if (idx >= nout * DMP_) return;
    const int k = idx % DMP_;
    const int o = idx / DMP_;
    Wt[idx] = (k < DM_) ? (__bf16)W[(size_t)k * nout + o] : (__bf16)0.0f;
}

__global__ void zero_f32(float* __restrict__ p, int n)
{
    int i = blockIdx.x * blockDim.x + threadIdx.x;
    if (i < n) p[i] = 0.0f;
}

// ---------------------------------------------------------------------------
// Host orchestration
// ---------------------------------------------------------------------------
extern "C" void kernel_launch(void* const* d_in, const int* in_sizes, int n_in,
                              void* d_out, int out_size, void* d_ws, size_t ws_size,
                              hipStream_t stream)
{
    const float* inputs  = (const float*)d_in[0];
    const float* graph   = (const float*)d_in[1];
    const float* eWru    = (const float*)d_in[2];
    const float* ebru    = (const float*)d_in[3];
    const float* eWc     = (const float*)d_in[4];
    const float* ebc     = (const float*)d_in[5];
    const float* dWru    = (const float*)d_in[6];
    const float* dbru    = (const float*)d_in[7];
    const float* dWc     = (const float*)d_in[8];
    const float* dbc     = (const float*)d_in[9];
    const float* dWo     = (const float*)d_in[10];
    const float* dbo     = (const float*)d_in[11];
    float* dout = (float*)d_out;

    // Carve workspace (256B aligned slices)
    char* base = (char*)d_ws;
    size_t off = 0;
    auto alloc = [&](size_t bytes) -> char* {
        char* p = base + off;
        off += (bytes + 255) & ~(size_t)255;
        return p;
    };
    __bf16* Abf    = (__bf16*)alloc((size_t)N_ * N_ * sizeof(__bf16));
    __bf16* Wru_e  = (__bf16*)alloc((size_t)128 * DMP_ * sizeof(__bf16));
    __bf16* Wc_e   = (__bf16*)alloc((size_t)64  * DMP_ * sizeof(__bf16));
    __bf16* Wru_d  = (__bf16*)alloc((size_t)128 * DMP_ * sizeof(__bf16));
    __bf16* Wc_d   = (__bf16*)alloc((size_t)64  * DMP_ * sizeof(__bf16));
    float*  XPAN   = (float*) alloc((size_t)4 * C_ * N_ * sizeof(float));
    __bf16* XPANbf = (__bf16*)alloc((size_t)4 * C_ * N_ * sizeof(__bf16));
    __bf16* XK     = (__bf16*)alloc((size_t)R_ * DMP_ * sizeof(__bf16));
    float*  RUt    = (float*) alloc((size_t)128 * R_ * sizeof(float));
    float*  Cct    = (float*) alloc((size_t)64  * R_ * sizeof(float));
    float*  Ht     = (float*) alloc((size_t)H_  * R_ * sizeof(float));
    float*  dec_in = (float*) alloc((size_t)R_ * sizeof(float));
    (void)ws_size; (void)in_sizes; (void)n_in; (void)out_size;

    auto XP  = [&](int m) { return XPAN   + (size_t)m * C_ * N_; };
    auto XPb = [&](int m) { return XPANbf + (size_t)m * C_ * N_; };

    const dim3 wblk(32, 4);

    // One-time-per-call conversions
    convert_graph<<<(N_ * N_ + 255) / 256, 256, 0, stream>>>(graph, Abf, N_ * N_);
    convert_w<<<(128 * DMP_ + 255) / 256, 256, 0, stream>>>(eWru, Wru_e, 128);
    convert_w<<<(64  * DMP_ + 255) / 256, 256, 0, stream>>>(eWc,  Wc_e,  64);
    convert_w<<<(128 * DMP_ + 255) / 256, 256, 0, stream>>>(dWru, Wru_d, 128);
    convert_w<<<(64  * DMP_ + 255) / 256, 256, 0, stream>>>(dWc,  Wc_d,  64);
    zero_f32<<<(H_ * R_ + 255) / 256, 256, 0, stream>>>(Ht, H_ * R_);

    auto cheb_chain = [&]() {
        // X1 = A @ X0 ; X2 = 2(A@X1) - X0 ; X3 = 2(A@X2) - X1
        cheb_gemm<<<dim3(4, 65), wblk, 0, stream>>>(Abf, XPb(0), XP(0), 1.0f,  0.0f, XP(1), XPb(1));
        cheb_gemm<<<dim3(4, 65), wblk, 0, stream>>>(Abf, XPb(1), XP(0), 2.0f, -1.0f, XP(2), XPb(2));
        cheb_gemm<<<dim3(4, 65), wblk, 0, stream>>>(Abf, XPb(2), XP(1), 2.0f, -1.0f, XP(3), XPb(3));
    };

    auto run_cell = [&](const float* xsrc, int xstrideB,
                        const __bf16* WruT, const float* bru,
                        const __bf16* WcT,  const float* bc) {
        const int packN = (C_ * N_ + 255) / 256;
        const int gathN = (R_ * DMP_ + 255) / 256;
        // gconv for r,u gates
        pack_x<<<packN, 256, 0, stream>>>(xsrc, xstrideB, Ht, nullptr, nullptr, XP(0), XPb(0));
        cheb_chain();
        gather_xk<<<gathN, 256, 0, stream>>>(XPAN, XK);
        lin_gemm<<<dim3(256, 8), wblk, 0, stream>>>(XK, WruT, RUt);
        // gconv for candidate c (input gated by r)
        pack_x<<<packN, 256, 0, stream>>>(xsrc, xstrideB, Ht, RUt, bru, XP(0), XPb(0));
        cheb_chain();
        gather_xk<<<gathN, 256, 0, stream>>>(XPAN, XK);
        lin_gemm<<<dim3(256, 4), wblk, 0, stream>>>(XK, WcT, Cct);
        // h = u*h + (1-u)*tanh(c)
        update_h<<<(H_ * R_ + 255) / 256, 256, 0, stream>>>(Ht, RUt, bru, Cct, bc);
    };

    // Encoder: 12 steps over inputs [B, T, N, 1]
    for (int t = 0; t < T_; ++t)
        run_cell(inputs + (size_t)t * N_, T_ * N_, Wru_e, ebru, Wc_e, ebc);

    // Decoder: autoregressive, first input is zeros
    zero_f32<<<(R_ + 255) / 256, 256, 0, stream>>>(dec_in, R_);
    for (int t = 0; t < NPRED_; ++t) {
        run_cell(dec_in, N_, Wru_d, dbru, Wc_d, dbc);
        project_out<<<(R_ + 255) / 256, 256, 0, stream>>>(Ht, dWo, dbo, dec_in, dout, t);
    }
}
